// XLNet_74328704025035
// MI455X (gfx1250) — matrix operand order repaired and checked by reading, not compiled, not run on hardware
//
#include <hip/hip_runtime.h>
#include <hip/hip_bf16.h>

#define DM   1024
#define NH   16
#define DH   64
#define DI   4096
#define NL   4
#define NT   32000
#define QL   1024
#define BS   2
#define MTOK (QL*BS)   /* 2048 rows (i,b) */
#define RL   (2*QL)    /* 2048 relative positions */

#define KT    64       /* K-tile staged in LDS per pipeline stage */
#define LDS_S 72       /* padded LDS row stride (elements): 144B, destaggers banks */

typedef __bf16 bf16;
typedef __attribute__((ext_vector_type(16))) __bf16 v16bf;
typedef __attribute__((ext_vector_type(8)))  float  v8f;
typedef __attribute__((ext_vector_type(4)))  unsigned int v4u;

union BFrag { v16bf v; v4u q[2]; };

__device__ inline bf16 to_bf16(float x) {
  union { float f; unsigned u; } a; a.f = x;
  unsigned r = a.u + 0x7FFFu + ((a.u >> 16) & 1u);   // round-to-nearest-even
  union { unsigned short s; bf16 b; } o;
  o.s = (unsigned short)(r >> 16);
  return o.b;
}

// Async 16-byte global -> LDS copy (CDNA5 async path, ASYNCcnt-tracked).
// LDS address = low 32 bits of the generic pointer (LDS aperture, ISA 10.2).
__device__ __forceinline__ void async_copy_b128(void* lds_dst, const void* gsrc) {
  unsigned lds_off = (unsigned)(unsigned long long)lds_dst;
  asm volatile("global_load_async_to_lds_b128 %0, %1, off"
               :: "v"(lds_off), "v"(gsrc) : "memory");
}
#define WAIT_ASYNC(imm) asm volatile("s_wait_asynccnt " #imm ::: "memory")

// ---------------------------------------------------------------------------
// Generic batched GEMM:  C[M,N] (f32) = A[M,K] (bf16, row-major, lda)
//                                     x Bt[N,K] (bf16, row-major, ldb)  (+ bias[n])
// Two-level batch: z -> (b1,b2) with element strides sX1/sX2.
// Block: 256 threads = 8 waves; block tile 128x64; K staged 64 at a time in
// double-buffered LDS via global_load_async_to_lds_b128; each wave computes a
// 16x64 strip with 4 f32 accumulators (v_wmma_f32_16x16x32_bf16).
// cmode: 1 = skip block if n0 >= mblk+128  (causal scores; masked anyway)
//        2 = skip block if n0 >= QL+64     (bd cols feeding only masked j)
// kcap:  1 = cap K at mblk+128             (probs are exactly 0 for j > i)
// Grid: (N/64, M/128, nbatch).  All dims divide exactly for this model.
// ---------------------------------------------------------------------------
__global__ __launch_bounds__(256)
void gemm_bf16_tn(const bf16* __restrict__ A, const bf16* __restrict__ Bt,
                  float* __restrict__ C, const float* __restrict__ bias,
                  int M, int N, int K, int lda, int ldb, int ldc, int nb2,
                  long sA1, long sA2, long sB1, long sB2, long sC1, long sC2,
                  int cmode, int kcap)
{
  __shared__ __align__(16) bf16 lA[2][128 * LDS_S];
  __shared__ __align__(16) bf16 lB[2][64 * LDS_S];

  const int mblk = blockIdx.y * 128;
  const int n0   = blockIdx.x * 64;
  if (cmode == 1 && n0 >= mblk + 128) return;   // uniform across block
  if (cmode == 2 && n0 >= QL + 64)    return;
  if (kcap) K = min(K, mblk + 128);

  int bb = blockIdx.z;
  int b1 = bb / nb2, b2 = bb % nb2;
  A  += b1 * sA1 + b2 * sA2;
  Bt += b1 * sB1 + b2 * sB2;
  C  += b1 * sC1 + b2 * sC2;

  const int lane = threadIdx.x;                 // 0..31
  const int wave = threadIdx.y;                 // 0..7
  const int tid  = wave * 32 + lane;            // 0..255

  // Cooperative copy mapping: 16B chunk = 8 bf16.
  const int crow = tid >> 3;                    // 0..31
  const int ccol = (tid & 7) * 8;               // element col offset, 0..56

  const bf16* agbase = A + (long)mblk * lda + ccol;
  const bf16* bgbase = Bt + (long)n0 * ldb + ccol;

  const int nk = K / KT;

  // issue one K-tile (6 async b128 per thread: 4 for A 128x64, 2 for B 64x64)
  auto issue = [&](int kt, int sel) {
    long k = (long)kt * KT;
#pragma unroll
    for (int i = 0; i < 4; ++i) {
      int r = crow + 32 * i;
      async_copy_b128(&lA[sel][r * LDS_S + ccol], agbase + (long)r * lda + k);
    }
#pragma unroll
    for (int i = 0; i < 2; ++i) {
      int r = crow + 32 * i;
      async_copy_b128(&lB[sel][r * LDS_S + ccol], bgbase + (long)r * ldb + k);
    }
  };

  v8f zero = {};
  v8f acc[4] = {zero, zero, zero, zero};

  // WMMA fragment addressing (ISA 16-bit A 16x32 / B 32x16 layouts):
  const int ar   = lane & 15;
  const int asel = (lane >> 4) ? 8 : 0;   // A: lanes16-31 hold K {8..15,24..31}
  const int bn   = lane & 15;
  const int bsel = (lane >> 4) ? 8 : 0;   // B: lanes16-31 hold K {8..15,24..31}

  issue(0, 0);
  for (int kt = 0; kt < nk; ++kt) {
    int sel = kt & 1;
    __syncthreads();                       // prior tile's LDS reads complete
    if (kt + 1 < nk) {
      issue(kt + 1, sel ^ 1);              // overlap next loads with compute
      WAIT_ASYNC(6);                       // older 6 (= tile kt) retired
    } else {
      WAIT_ASYNC(0);
    }
    __syncthreads();                       // tile kt visible from all waves

#pragma unroll
    for (int k0 = 0; k0 < KT; k0 += 32) {
      BFrag a;
      a.q[0] = *(const v4u*)&lA[sel][(wave * 16 + ar) * LDS_S + k0 + asel];
      a.q[1] = *(const v4u*)&lA[sel][(wave * 16 + ar) * LDS_S + k0 + asel + 16];
#pragma unroll
      for (int t = 0; t < 4; ++t) {
        BFrag b;
        b.q[0] = *(const v4u*)&lB[sel][(t * 16 + bn) * LDS_S + k0 + bsel];
        b.q[1] = *(const v4u*)&lB[sel][(t * 16 + bn) * LDS_S + k0 + bsel + 16];
        acc[t] = __builtin_amdgcn_wmma_f32_16x16x32_bf16(
            false, a.v, false, b.v, (short)0, acc[t], false, false);
      }
    }
  }

  // C layout: lanes 0-15 rows 0..7 (vgpr idx), lanes 16-31 rows 8..15.
  const int col   = lane & 15;
  const int rbase = (lane >> 4) * 8;
  const int m0    = mblk + wave * 16;
#pragma unroll
  for (int t = 0; t < 4; ++t) {
    int nc = n0 + t * 16 + col;
    float bv = bias ? bias[nc] : 0.0f;
#pragma unroll
    for (int r = 0; r < 8; ++r)
      C[(long)(m0 + rbase + r) * ldc + nc] = acc[t][r] + bv;
  }
}

// ---------------------------------------------------------------------------
// Elementwise / support kernels
// ---------------------------------------------------------------------------
__global__ __launch_bounds__(256)
void embed_gather(const int* __restrict__ ids, const float* __restrict__ emb,
                  float* __restrict__ h, bf16* __restrict__ hbf)
{
  long idx = (long)blockIdx.x * 256 + threadIdx.x;   // MTOK*DM
  int c = idx & (DM - 1);
  int m = (int)(idx >> 10);
  float v = emb[(long)ids[m] * DM + c];
  h[idx] = v;
  hbf[idx] = to_bf16(v);
}

__global__ __launch_bounds__(256)
void transpose_convert(const float* __restrict__ in, bf16* __restrict__ out,
                       int K, int N)
{
  long idx = (long)blockIdx.x * 256 + threadIdx.x;   // over N*K outputs
  if (idx >= (long)K * N) return;
  long nn = idx / K, kk = idx % K;
  out[idx] = to_bf16(in[kk * (long)N + nn]);
}

__global__ __launch_bounds__(256)
void bias_convert(const float* __restrict__ in, const float* __restrict__ bias,
                  bf16* __restrict__ out, long n, int ncol)
{
  long idx = (long)blockIdx.x * 256 + threadIdx.x;
  if (idx >= n) return;
  float v = in[idx];
  if (bias) v += bias[idx % ncol];
  out[idx] = to_bf16(v);
}

__global__ __launch_bounds__(256)
void pos_emb_kernel(bf16* __restrict__ out)
{
  long idx = (long)blockIdx.x * 256 + threadIdx.x;   // RL*DM
  int c = idx & (DM - 1);
  int m = (int)(idx >> 10);
  float pos = (float)(QL - m);                        // pos_seq[m] = klen - m
  int ci = c & 511;
  float freq = __expf(-9.210340371976184f * (float)(2 * ci) * (1.0f / DM));
  float ang = pos * freq;
  out[idx] = to_bf16((c < 512) ? __sinf(ang) : __cosf(ang));
}

__global__ __launch_bounds__(256)
void vtranspose(const float* __restrict__ v, bf16* __restrict__ out)
{
  int tid = blockIdx.x * 256 + threadIdx.x;           // BS*NH*DH*QL
  int j = tid & (QL - 1);
  int d = (tid >> 10) & (DH - 1);
  int n = (tid >> 16) & (NH - 1);
  int b = tid >> 20;
  out[tid] = to_bf16(v[(long)(j * BS + b) * DM + n * DH + d]);
}

// ef[i,b,n,s] = sum_d (q[i,b,n,d]+r_s_bias[n,d]) * seg_embed[s,n,d]
__global__ __launch_bounds__(256)
void ef_kernel(const float* __restrict__ q, const float* __restrict__ rs,
               const float* __restrict__ se, float* __restrict__ ef)
{
  int tid = blockIdx.x * 256 + threadIdx.x;           // QL*BS*NH*2
  int s = tid & 1;
  int n = (tid >> 1) & (NH - 1);
  int b = (tid >> 5) & 1;
  int i = tid >> 6;
  const float* qp = q + (long)(i * BS + b) * DM + n * DH;
  const float* bp = rs + n * DH;
  const float* sp = se + (s * NH + n) * DH;
  float acc = 0.0f;
#pragma unroll 8
  for (int d = 0; d < DH; ++d) acc += (qp[d] + bp[d]) * sp[d];
  ef[tid] = acc;   // tid == ((i*BS+b)*NH+n)*2+s by construction
}

// Per attention row: score = (ac + bd_shifted + ef_sel)*scale, causal mask,
// stable softmax, write bf16 probs.  rel_shift: bd[i,j] = bd_raw[i, QL - i + j].
__global__ __launch_bounds__(256)
void score_softmax(const float* __restrict__ ac, const float* __restrict__ bd,
                   const float* __restrict__ ef, const int* __restrict__ seg,
                   bf16* __restrict__ probs)
{
  int i  = blockIdx.x;                // query position
  int bb = blockIdx.y;                // b*NH + n
  int b = bb >> 4, n = bb & 15;
  const float scale = 0.125f;         // 1/sqrt(64)
  long base  = (long)bb * (QL * QL) + (long)i * QL;
  long bbase = (long)bb * (QL * RL) + (long)i * RL + (QL - i);
  int  segi  = seg[i * BS + b];
  const float* efp = ef + ((long)(i * BS + b) * NH + n) * 2;
  float ef0 = efp[0], ef1 = efp[1];

  __shared__ float red[256];
  int t = threadIdx.x;
  float s[4];
  float mx = -3.0e38f;
#pragma unroll
  for (int u = 0; u < 4; ++u) {
    int j = t + u * 256;
    int segj = seg[j * BS + b];
    float e = (segj != segi) ? ef1 : ef0;
    float v = (ac[base + j] + bd[bbase + j] + e) * scale;
    if (j > i) v = -1.0e30f;
    s[u] = v;
    mx = fmaxf(mx, v);
  }
  red[t] = mx; __syncthreads();
  for (int st = 128; st > 0; st >>= 1) {
    if (t < st) red[t] = fmaxf(red[t], red[t + st]);
    __syncthreads();
  }
  mx = red[0]; __syncthreads();

  float sum = 0.0f;
#pragma unroll
  for (int u = 0; u < 4; ++u) { s[u] = __expf(s[u] - mx); sum += s[u]; }
  red[t] = sum; __syncthreads();
  for (int st = 128; st > 0; st >>= 1) {
    if (t < st) red[t] += red[t + st];
    __syncthreads();
  }
  float inv = 1.0f / red[0];
#pragma unroll
  for (int u = 0; u < 4; ++u) {
    int j = t + u * 256;
    probs[base + j] = to_bf16(s[u] * inv);
  }
}

__global__ __launch_bounds__(256)
void gelu_convert(const float* __restrict__ in, bf16* __restrict__ out, long n)
{
  long idx = (long)blockIdx.x * 256 + threadIdx.x;
  if (idx >= n) return;
  float x = in[idx];
  out[idx] = to_bf16(0.5f * x * (1.0f + erff(x * 0.70710678118654752f)));
}

// h = LayerNorm(x + h) * g + b ; also writes bf16 copy of h.
__global__ __launch_bounds__(256)
void residual_ln(const float* __restrict__ x, float* __restrict__ h,
                 bf16* __restrict__ hbf, const float* __restrict__ g,
                 const float* __restrict__ be)
{
  int m = blockIdx.x, t = threadIdx.x;
  __shared__ float red[256];
  float v[4]; float s = 0.0f;
#pragma unroll
  for (int u = 0; u < 4; ++u) {
    int c = t + u * 256;
    v[u] = x[(long)m * DM + c] + h[(long)m * DM + c];
    s += v[u];
  }
  red[t] = s; __syncthreads();
  for (int st = 128; st > 0; st >>= 1) { if (t < st) red[t] += red[t + st]; __syncthreads(); }
  float mu = red[0] * (1.0f / DM); __syncthreads();
  float s2 = 0.0f;
#pragma unroll
  for (int u = 0; u < 4; ++u) { v[u] -= mu; s2 += v[u] * v[u]; }
  red[t] = s2; __syncthreads();
  for (int st = 128; st > 0; st >>= 1) { if (t < st) red[t] += red[t + st]; __syncthreads(); }
  float rstd = rsqrtf(red[0] * (1.0f / DM) + 1e-5f);
#pragma unroll
  for (int u = 0; u < 4; ++u) {
    int c = t + u * 256;
    float o = v[u] * rstd * g[c] + be[c];
    h[(long)m * DM + c]  = o;
    hbf[(long)m * DM + c] = to_bf16(o);
  }
}

// ---------------------------------------------------------------------------
extern "C" void kernel_launch(void* const* d_in, const int* in_sizes, int n_in,
                              void* d_out, int out_size, void* d_ws, size_t ws_size,
                              hipStream_t stream)
{
  (void)in_sizes; (void)n_in; (void)out_size; (void)ws_size;
  const int*   inp_k     = (const int*)  d_in[0];
  const int*   seg_id    = (const int*)  d_in[1];
  const float* emb       = (const float*)d_in[2];
  const float* r_w_bias  = (const float*)d_in[3];
  const float* r_r_bias  = (const float*)d_in[4];
  const float* r_s_bias  = (const float*)d_in[5];
  const float* seg_embed = (const float*)d_in[6];
  const float* q_proj    = (const float*)d_in[7];
  const float* k_proj    = (const float*)d_in[8];
  const float* v_proj    = (const float*)d_in[9];
  const float* r_proj    = (const float*)d_in[10];
  const float* proj_o    = (const float*)d_in[11];
  const float* ln_g      = (const float*)d_in[12];
  const float* ln_b      = (const float*)d_in[13];
  const float* w1        = (const float*)d_in[14];
  const float* b1        = (const float*)d_in[15];
  const float* w2        = (const float*)d_in[16];
  const float* b2        = (const float*)d_in[17];
  const float* softmax_b = (const float*)d_in[18];

  char* wsb = (char*)d_ws;
  size_t off = 0;
  auto alloc = [&](size_t bytes) -> char* {
    char* p = wsb + off;
    off = (off + bytes + 255) & ~(size_t)255;
    return p;
  };

  bf16*  emb_bf = (bf16*)alloc((size_t)NT * DM * 2);
  bf16*  wq_t   = (bf16*)alloc((size_t)DM * DM * 2);
  bf16*  wk_t   = (bf16*)alloc((size_t)DM * DM * 2);
  bf16*  wv_t   = (bf16*)alloc((size_t)DM * DM * 2);
  bf16*  wr_t   = (bf16*)alloc((size_t)DM * DM * 2);
  bf16*  wo_bf  = (bf16*)alloc((size_t)DM * DM * 2);
  bf16*  w1_t   = (bf16*)alloc((size_t)DM * DI * 2);
  bf16*  w2_t   = (bf16*)alloc((size_t)DI * DM * 2);
  bf16*  pos_bf = (bf16*)alloc((size_t)RL * DM * 2);
  float* kr_f   = (float*)alloc((size_t)RL * DM * 4);
  bf16*  kr_bf  = (bf16*)alloc((size_t)RL * DM * 2);
  float* h_f    = (float*)alloc((size_t)MTOK * DM * 4);
  bf16*  h_bf   = (bf16*)alloc((size_t)MTOK * DM * 2);
  float* q_f    = (float*)alloc((size_t)MTOK * DM * 4);
  float* k_f    = (float*)alloc((size_t)MTOK * DM * 4);
  float* v_f    = (float*)alloc((size_t)MTOK * DM * 4);
  bf16*  qw_bf  = (bf16*)alloc((size_t)MTOK * DM * 2);
  bf16*  qr_bf  = (bf16*)alloc((size_t)MTOK * DM * 2);
  bf16*  k_bf   = (bf16*)alloc((size_t)MTOK * DM * 2);
  bf16*  vT_bf  = (bf16*)alloc((size_t)BS * NH * DH * QL * 2);
  float* ef_f   = (float*)alloc((size_t)QL * BS * NH * 2 * 4);
  float* ac_f   = (float*)alloc((size_t)BS * NH * QL * QL * 4);
  float* bd_f   = (float*)alloc((size_t)BS * NH * QL * RL * 4);
  bf16*  probs  = (bf16*)alloc((size_t)BS * NH * QL * QL * 2);
  float* av_f   = (float*)alloc((size_t)MTOK * DM * 4);
  bf16*  av_bf  = (bf16*)alloc((size_t)MTOK * DM * 2);
  float* ao_f   = (float*)alloc((size_t)MTOK * DM * 4);
  float* ff1_f  = (float*)alloc((size_t)MTOK * DI * 4);
  bf16*  ff1_bf = (bf16*)alloc((size_t)MTOK * DI * 2);
  float* ff2_f  = (float*)alloc((size_t)MTOK * DM * 4);

  auto gemm = [&](const bf16* A, const bf16* Bt, float* C, const float* bias,
                  int M, int N, int K, int lda, int ldb, int ldc,
                  int nb, int nb2, long sA1, long sA2, long sB1, long sB2,
                  long sC1, long sC2, int cmode, int kcap) {
    dim3 grid(N / 64, M / 128, nb);
    dim3 blk(32, 8);
    gemm_bf16_tn<<<grid, blk, 0, stream>>>(A, Bt, C, bias, M, N, K, lda, ldb,
                                           ldc, nb2, sA1, sA2, sB1, sB2, sC1, sC2,
                                           cmode, kcap);
  };
  auto cvt = [&](const float* in, const float* bias, bf16* out, long n, int ncol) {
    bias_convert<<<(int)((n + 255) / 256), 256, 0, stream>>>(in, bias, out, n, ncol);
  };

  // ---- one-time weight prep ------------------------------------------------
  cvt(emb, nullptr, emb_bf, (long)NT * DM, DM);
  transpose_convert<<<(DM * DM) / 256, 256, 0, stream>>>(q_proj, wq_t, DM, DM);
  transpose_convert<<<(DM * DM) / 256, 256, 0, stream>>>(k_proj, wk_t, DM, DM);
  transpose_convert<<<(DM * DM) / 256, 256, 0, stream>>>(v_proj, wv_t, DM, DM);
  transpose_convert<<<(DM * DM) / 256, 256, 0, stream>>>(r_proj, wr_t, DM, DM);
  cvt(proj_o, nullptr, wo_bf, (long)DM * DM, DM);           // Bt for proj_o is proj_o itself
  transpose_convert<<<(DM * DI) / 256, 256, 0, stream>>>(w1, w1_t, DM, DI);
  transpose_convert<<<(DI * DM) / 256, 256, 0, stream>>>(w2, w2_t, DI, DM);

  // ---- positional encoding: k_head_r = pos_emb @ r_proj --------------------
  pos_emb_kernel<<<(RL * DM) / 256, 256, 0, stream>>>(pos_bf);
  gemm(pos_bf, wr_t, kr_f, nullptr, RL, DM, DM, DM, DM, DM, 1, 1,
       0,0,0,0,0,0, 0, 0);
  cvt(kr_f, nullptr, kr_bf, (long)RL * DM, DM);

  // ---- embedding gather ----------------------------------------------------
  embed_gather<<<(MTOK * DM) / 256, 256, 0, stream>>>(inp_k, emb, h_f, h_bf);

  // ---- transformer layers --------------------------------------------------
  for (int L = 0; L < NL; ++L) {
    const float* rw = r_w_bias + (long)L * NH * DH;
    const float* rr = r_r_bias + (long)L * NH * DH;
    const float* rs = r_s_bias + (long)L * NH * DH;
    const float* se = seg_embed + (long)L * 2 * NH * DH;

    gemm(h_bf, wq_t, q_f, nullptr, MTOK, DM, DM, DM, DM, DM, 1, 1,
         0,0,0,0,0,0, 0, 0);
    gemm(h_bf, wk_t, k_f, nullptr, MTOK, DM, DM, DM, DM, DM, 1, 1,
         0,0,0,0,0,0, 0, 0);
    gemm(h_bf, wv_t, v_f, nullptr, MTOK, DM, DM, DM, DM, DM, 1, 1,
         0,0,0,0,0,0, 0, 0);

    cvt(q_f, rw, qw_bf, (long)MTOK * DM, DM);
    cvt(q_f, rr, qr_bf, (long)MTOK * DM, DM);
    cvt(k_f, nullptr, k_bf, (long)MTOK * DM, DM);
    vtranspose<<<(BS * NH * DH * QL) / 256, 256, 0, stream>>>(v_f, vT_bf);
    ef_kernel<<<(QL * BS * NH * 2) / 256, 256, 0, stream>>>(q_f, rs, se, ef_f);

    // ac[b,n,i,j] = (q + r_w_bias) . k   batched over (b,n); causal blocks skipped
    gemm(qw_bf, k_bf, ac_f, nullptr, QL, QL, DH, BS*DM, BS*DM, QL,
         BS * NH, NH, DM, DH, DM, DH, (long)NH * QL * QL, (long)QL * QL, 1, 0);
    // bd_raw[b,n,i,m] = (q + r_r_bias) . k_head_r; only m <= QL feeds unmasked j
    gemm(qr_bf, kr_bf, bd_f, nullptr, QL, RL, DH, BS*DM, DM, RL,
         BS * NH, NH, DM, DH, 0, DH, (long)NH * QL * RL, (long)QL * RL, 2, 0);

    score_softmax<<<dim3(QL, BS * NH), 256, 0, stream>>>(ac_f, bd_f, ef_f,
                                                         seg_id, probs);

    // attn_vec[i,b,n,d] = probs . v ; probs[i,j>i]==0 -> K capped at mblk+128
    gemm(probs, vT_bf, av_f, nullptr, QL, DH, QL, QL, QL, BS*DM,
         BS * NH, NH, (long)NH * QL * QL, (long)QL * QL,
         (long)NH * DH * QL, (long)DH * QL, (long)DM, (long)DH, 0, 1);

    cvt(av_f, nullptr, av_bf, (long)MTOK * DM, DM);
    gemm(av_bf, wo_bf, ao_f, nullptr, MTOK, DM, DM, DM, DM, DM, 1, 1,
         0,0,0,0,0,0, 0, 0);
    residual_ln<<<MTOK, 256, 0, stream>>>(ao_f, h_f, h_bf, ln_g, ln_b);

    // FFN
    gemm(h_bf, w1_t, ff1_f, b1, MTOK, DI, DM, DM, DM, DI, 1, 1,
         0,0,0,0,0,0, 0, 0);
    gelu_convert<<<(MTOK * DI) / 256, 256, 0, stream>>>(ff1_f, ff1_bf, (long)MTOK * DI);
    gemm(ff1_bf, w2_t, ff2_f, b2, MTOK, DM, DI, DI, DI, DM, 1, 1,
         0,0,0,0,0,0, 0, 0);
    residual_ln<<<MTOK, 256, 0, stream>>>(ff2_f, h_f, h_bf, ln_g, ln_b);
  }

  // ---- LM head: logits = h @ emb^T + softmax_b -----------------------------
  gemm(h_bf, emb_bf, (float*)d_out, softmax_b, MTOK, NT, DM, DM, DM, NT,
       1, 1, 0, 0, 0, 0, 0, 0, 0, 0);
}
